// ResidualAttentionBlock_60438779789851
// MI455X (gfx1250) — compile-verified, matrix operand order checked by reading, not run on hardware
//
#include <hip/hip_runtime.h>
#include <math.h>

// ---------------- problem constants (match reference) ----------------
#define L_SEQ  2048
#define NB     2
#define EMB    1024
#define NHEAD  16
#define HDIM   64
#define MROWS  (L_SEQ * NB)          // 4096 (l,n) rows
#define QSCALE (1.0f / 64.0f)        // head_dim**-1 * sqrt(attn_mult)

// ---------------- WMMA fragment types ----------------
typedef __attribute__((ext_vector_type(16))) __bf16       bf16x16;
typedef __attribute__((ext_vector_type(8)))  float        f32x8;
typedef __attribute__((ext_vector_type(4)))  unsigned int u32x4;

union AFrag {
    bf16x16 v;
    u32x4   q[2];   // two 16B chunks (8 bf16 each)
    __bf16  e[16];
};

static __device__ __forceinline__ f32x8 wmma_bf16(bf16x16 a, bf16x16 b, f32x8 c) {
    // D(16x16 f32) = A(16x32 bf16) * B(32x16 bf16) + C
    return __builtin_amdgcn_wmma_f32_16x16x32_bf16(
        /*neg_a=*/false, a, /*neg_b=*/false, b,
        /*c_mod=*/(short)0, c, /*reuse_a=*/false, /*reuse_b=*/false);
}

// ---------------- fp32 -> bf16 weight conversion ----------------
__global__ __launch_bounds__(256)
void f32_to_bf16_k(const float* __restrict__ in, __bf16* __restrict__ out, int n) {
    int i = blockIdx.x * 256 + threadIdx.x;
    if (i < n) out[i] = (__bf16)in[i];
}

// ---------------- LayerNorm -> bf16 (one block per row of 1024) ----------------
__global__ __launch_bounds__(256)
void ln_to_bf16_k(const float* __restrict__ x, const float* __restrict__ w,
                  const float* __restrict__ b, __bf16* __restrict__ y) {
    __shared__ float rs[256], rq[256];
    const int row = blockIdx.x, tid = threadIdx.x;
    const float* xr = x + (size_t)row * EMB;
    const float4 f = *(const float4*)(xr + tid * 4);
    rs[tid] = f.x + f.y + f.z + f.w;
    rq[tid] = f.x * f.x + f.y * f.y + f.z * f.z + f.w * f.w;
    __syncthreads();
    for (int k = 128; k > 0; k >>= 1) {
        if (tid < k) { rs[tid] += rs[tid + k]; rq[tid] += rq[tid + k]; }
        __syncthreads();
    }
    const float mean = rs[0] * (1.0f / EMB);
    const float var  = rq[0] * (1.0f / EMB) - mean * mean;
    const float rstd = rsqrtf(var + 1e-5f);
    const int c0 = tid * 4;
    const float4 wv = *(const float4*)(w + c0);
    const float4 bv = *(const float4*)(b + c0);
    __bf16* yr = y + (size_t)row * EMB;
    yr[c0 + 0] = (__bf16)((f.x - mean) * rstd * wv.x + bv.x);
    yr[c0 + 1] = (__bf16)((f.y - mean) * rstd * wv.y + bv.y);
    yr[c0 + 2] = (__bf16)((f.z - mean) * rstd * wv.z + bv.z);
    yr[c0 + 3] = (__bf16)((f.w - mean) * rstd * wv.w + bv.w);
}

// ---------------- generic bf16 WMMA GEMM: C[M,Nc] = A[M,K] * W[Nc,K]^T + bias -------
// 256 threads = 8 waves; wave computes 32x64; block tile 256x64.
// Weight tile (64x32) staged once per block in LDS; A-frags straight from global.
template<bool GELU_ACT, bool RESID, bool OUT_BF16>
__global__ __launch_bounds__(256)
void gemm_bf16_k(const __bf16* __restrict__ A, const __bf16* __restrict__ W,
                 const float* __restrict__ bias, const float* __restrict__ res,
                 float* __restrict__ Cf, __bf16* __restrict__ Cb,
                 int M, int Nc, int K) {
    __shared__ __bf16 Bt[64][40];        // 40-elem row stride breaks bank conflicts
    const int tid    = threadIdx.x;
    const int lid    = tid & 31;
    const int wid    = tid >> 5;
    const int colA   = lid & 15;        // A-frag row / B-frag col / C-frag col
    const int halfid = lid >> 4;        // which 16-lane half
    const int kb     = halfid * 8;      // k-chunk base within fragment
    const int m0 = blockIdx.y * 256 + wid * 32;
    const int n0 = blockIdx.x * 64;
    const int brow = tid >> 2;           // cooperative B loader: 64 rows x 4 chunks
    const int bch  = (tid & 3) * 8;

    f32x8 acc0[4] = {}, acc1[4] = {};
    const __bf16* ar0  = A + (size_t)(m0 + colA) * K;
    const __bf16* ar1  = A + (size_t)(m0 + 16 + colA) * K;
    const __bf16* wsrc = W + (size_t)(n0 + brow) * K + bch;

    for (int k0 = 0; k0 < K; k0 += 32) {
        // stage 64x32 weight tile (4KB) into LDS once per block
        *(u32x4*)&Bt[brow][bch] = *(const u32x4*)(wsrc + k0);
        __syncthreads();
        AFrag a0, a1;
        a0.q[0] = *(const u32x4*)(ar0 + k0 + kb);
        a0.q[1] = *(const u32x4*)(ar0 + k0 + 16 + kb);
        a1.q[0] = *(const u32x4*)(ar1 + k0 + kb);
        a1.q[1] = *(const u32x4*)(ar1 + k0 + 16 + kb);
#pragma unroll
        for (int j = 0; j < 4; ++j) {
            AFrag b;
            b.q[0] = *(const u32x4*)&Bt[j * 16 + colA][kb];
            b.q[1] = *(const u32x4*)&Bt[j * 16 + colA][16 + kb];
            acc0[j] = wmma_bf16(a0.v, b.v, acc0[j]);
            acc1[j] = wmma_bf16(a1.v, b.v, acc1[j]);
        }
        __syncthreads();
    }
#pragma unroll
    for (int j = 0; j < 4; ++j) {
        const int col = n0 + j * 16 + colA;
        const float bv = bias[col];
#pragma unroll
        for (int r = 0; r < 8; ++r) {
            {
                const int m = m0 + r + 8 * halfid;
                const size_t idx = (size_t)m * Nc + col;
                float v = acc0[j][r] + bv;
                if (GELU_ACT) v = 0.5f * v * (1.0f + erff(v * 0.70710678118654752f));
                if (RESID)    v += res[idx];
                if (OUT_BF16) Cb[idx] = (__bf16)v; else Cf[idx] = v;
            }
            {
                const int m = m0 + 16 + r + 8 * halfid;
                const size_t idx = (size_t)m * Nc + col;
                float v = acc1[j][r] + bv;
                if (GELU_ACT) v = 0.5f * v * (1.0f + erff(v * 0.70710678118654752f));
                if (RESID)    v += res[idx];
                if (OUT_BF16) Cb[idx] = (__bf16)v; else Cf[idx] = v;
            }
        }
    }
}

// ---------------- qkv (fp32, (L*N,3E)) -> head layouts (bf16) ----------------
// q,k: (N,H,L,D); v transposed: (N,H,D,L) so PV B-fragments load contiguously.
__global__ __launch_bounds__(256)
void pack_qkv_k(const float* __restrict__ qkv, __bf16* __restrict__ q_p,
                __bf16* __restrict__ k_p, __bf16* __restrict__ v_t) {
    const int r = blockIdx.x;            // (l,n) row: r = l*NB + n
    const int l = r / NB, n = r % NB;
    const float* row = qkv + (size_t)r * (3 * EMB);
    for (int c = threadIdx.x; c < EMB; c += 256) {
        const int h = c >> 6, d = c & 63;
        const int head = n * NHEAD + h;
        q_p[((size_t)head * L_SEQ + l) * HDIM + d] = (__bf16)(row[c] * QSCALE);
        k_p[((size_t)head * L_SEQ + l) * HDIM + d] = (__bf16)(row[EMB + c]);
        v_t[((size_t)head * HDIM + d) * L_SEQ + l] = (__bf16)(row[2 * EMB + c]);
    }
}

// ---------------- flash attention (online softmax, WMMA) ----------------
// grid.x = N*H heads, grid.y = L/128; each wave owns 16 query rows.
__global__ __launch_bounds__(256)
void flash_attn_k(const __bf16* __restrict__ q_p, const __bf16* __restrict__ k_p,
                  const __bf16* __restrict__ v_t, __bf16* __restrict__ ob) {
    __shared__ __bf16 Pt[8][16][40];     // per-wave P transpose staging (padded)
    const int lid    = threadIdx.x & 31;
    const int wid    = threadIdx.x >> 5;
    const int colA   = lid & 15;
    const int halfid = lid >> 4;
    const int kb     = halfid * 8;
    const int head = blockIdx.x;         // head = n*NHEAD + h
    const int n = head >> 4;
    const int h = head & 15;
    const int m0 = blockIdx.y * 128 + wid * 16;

    const __bf16* qh = q_p + (size_t)head * L_SEQ * HDIM;
    const __bf16* kh = k_p + (size_t)head * L_SEQ * HDIM;
    const __bf16* vh = v_t + (size_t)head * HDIM * L_SEQ;

    // Q A-fragments for K-chunks d=[0,32) and [32,64)
    AFrag a0, a1;
    const __bf16* qrow = qh + (size_t)(m0 + colA) * HDIM;
    a0.q[0] = *(const u32x4*)(qrow + kb);
    a0.q[1] = *(const u32x4*)(qrow + 16 + kb);
    a1.q[0] = *(const u32x4*)(qrow + 32 + kb);
    a1.q[1] = *(const u32x4*)(qrow + 48 + kb);

    f32x8 o0 = {}, o1 = {}, o2 = {}, o3 = {};
    float mr[8], lr[8];
#pragma unroll
    for (int r = 0; r < 8; ++r) { mr[r] = -3.0e30f; lr[r] = 0.0f; }

    for (int s0 = 0; s0 < L_SEQ; s0 += 32) {
        // scores S = Q * K^T for 32 keys -> two 16x16 f32 tiles
        f32x8 c0 = {}, c1 = {};
        {
            const __bf16* krow = kh + (size_t)(s0 + colA) * HDIM;
            AFrag b;
            b.q[0] = *(const u32x4*)(krow + kb);
            b.q[1] = *(const u32x4*)(krow + 16 + kb);
            c0 = wmma_bf16(a0.v, b.v, c0);
            b.q[0] = *(const u32x4*)(krow + 32 + kb);
            b.q[1] = *(const u32x4*)(krow + 48 + kb);
            c0 = wmma_bf16(a1.v, b.v, c0);
            krow += 16 * HDIM;
            b.q[0] = *(const u32x4*)(krow + kb);
            b.q[1] = *(const u32x4*)(krow + 16 + kb);
            c1 = wmma_bf16(a0.v, b.v, c1);
            b.q[0] = *(const u32x4*)(krow + 32 + kb);
            b.q[1] = *(const u32x4*)(krow + 48 + kb);
            c1 = wmma_bf16(a1.v, b.v, c1);
        }
        // online softmax: rows live at fixed VGPR index across the 16-lane half,
        // so xor-shuffles 1,2,4,8 reduce over the key (column) dimension.
#pragma unroll
        for (int r = 0; r < 8; ++r) {
            float mx = fmaxf(c0[r], c1[r]);
            mx = fmaxf(mx, __shfl_xor(mx, 1, 32));
            mx = fmaxf(mx, __shfl_xor(mx, 2, 32));
            mx = fmaxf(mx, __shfl_xor(mx, 4, 32));
            mx = fmaxf(mx, __shfl_xor(mx, 8, 32));
            const float mnew  = fmaxf(mr[r], mx);
            const float scale = __expf(mr[r] - mnew);
            const float e0 = __expf(c0[r] - mnew);
            const float e1 = __expf(c1[r] - mnew);
            float sum = e0 + e1;
            sum += __shfl_xor(sum, 1, 32);
            sum += __shfl_xor(sum, 2, 32);
            sum += __shfl_xor(sum, 4, 32);
            sum += __shfl_xor(sum, 8, 32);
            lr[r] = lr[r] * scale + sum;
            mr[r] = mnew;
            c0[r] = e0; c1[r] = e1;
            o0[r] *= scale; o1[r] *= scale; o2[r] *= scale; o3[r] *= scale;
        }
        // transpose P (C layout -> A layout) through per-wave LDS tile
#pragma unroll
        for (int r = 0; r < 8; ++r) {
            Pt[wid][r + 8 * halfid][colA]      = (__bf16)c0[r];
            Pt[wid][r + 8 * halfid][16 + colA] = (__bf16)c1[r];
        }
        asm volatile("s_wait_dscnt 0x0" ::: "memory");  // same-wave cross-lane DS hazard
        AFrag p;
#pragma unroll
        for (int i = 0; i < 8; ++i) {
            p.e[i]     = Pt[wid][colA][kb + i];
            p.e[8 + i] = Pt[wid][colA][16 + kb + i];
        }
        asm volatile("s_wait_dscnt 0x0" ::: "memory");
        // O += P * V  (V stored transposed: contiguous along keys)
#pragma unroll
        for (int j = 0; j < 4; ++j) {
            const __bf16* vrow = vh + (size_t)(j * 16 + colA) * L_SEQ + s0;
            AFrag b;
            b.q[0] = *(const u32x4*)(vrow + kb);
            b.q[1] = *(const u32x4*)(vrow + 16 + kb);
            f32x8& oo = (j == 0) ? o0 : (j == 1) ? o1 : (j == 2) ? o2 : o3;
            oo = wmma_bf16(p.v, b.v, oo);
        }
    }
    // normalize and scatter to (L, N, E) bf16
    float inv[8];
#pragma unroll
    for (int r = 0; r < 8; ++r) inv[r] = 1.0f / lr[r];
#pragma unroll
    for (int j = 0; j < 4; ++j) {
        const f32x8& oo = (j == 0) ? o0 : (j == 1) ? o1 : (j == 2) ? o2 : o3;
#pragma unroll
        for (int r = 0; r < 8; ++r) {
            const int l_idx = m0 + r + 8 * halfid;
            const size_t idx = ((size_t)l_idx * NB + n) * EMB + h * HDIM + j * 16 + colA;
            ob[idx] = (__bf16)(oo[r] * inv[r]);
        }
    }
}

// ---------------- host-side orchestration ----------------
extern "C" void kernel_launch(void* const* d_in, const int* in_sizes, int n_in,
                              void* d_out, int out_size, void* d_ws, size_t ws_size,
                              hipStream_t stream) {
    const float* x         = (const float*)d_in[0];
    const float* ln1_w     = (const float*)d_in[1];
    const float* ln1_b     = (const float*)d_in[2];
    const float* in_proj_w = (const float*)d_in[3];
    const float* in_proj_b = (const float*)d_in[4];
    const float* out_w     = (const float*)d_in[5];
    const float* out_b     = (const float*)d_in[6];
    const float* ln2_w     = (const float*)d_in[7];
    const float* ln2_b     = (const float*)d_in[8];
    const float* fc_w      = (const float*)d_in[9];
    const float* fc_b      = (const float*)d_in[10];
    const float* proj_w    = (const float*)d_in[11];
    const float* proj_b    = (const float*)d_in[12];
    (void)in_sizes; (void)n_in; (void)out_size; (void)ws_size;

    char* ws = (char*)d_ws;
    size_t off = 0;
    auto take = [&](size_t bytes) -> void* {
        void* p = ws + off;
        off += (bytes + 255) & ~(size_t)255;
        return p;
    };
    __bf16* y_bf  = (__bf16*)take((size_t)MROWS * EMB * 2);
    __bf16* wqkv  = (__bf16*)take((size_t)3 * EMB * EMB * 2);
    __bf16* wout  = (__bf16*)take((size_t)EMB * EMB * 2);
    __bf16* wfc   = (__bf16*)take((size_t)4 * EMB * EMB * 2);
    __bf16* wproj = (__bf16*)take((size_t)4 * EMB * EMB * 2);
    float*  qkv_f = (float*) take((size_t)MROWS * 3 * EMB * 4);
    __bf16* q_p   = (__bf16*)take((size_t)NB * NHEAD * L_SEQ * HDIM * 2);
    __bf16* k_p   = (__bf16*)take((size_t)NB * NHEAD * L_SEQ * HDIM * 2);
    __bf16* v_t   = (__bf16*)take((size_t)NB * NHEAD * L_SEQ * HDIM * 2);
    __bf16* o_bf  = (__bf16*)take((size_t)MROWS * EMB * 2);
    float*  x1    = (float*) take((size_t)MROWS * EMB * 4);
    __bf16* z_bf  = (__bf16*)take((size_t)MROWS * EMB * 2);
    __bf16* h_bf  = (__bf16*)take((size_t)MROWS * 4 * EMB * 2);

    // weights -> bf16 (deterministic per call; cheap vs. GEMMs)
    int nc;
    nc = 3 * EMB * EMB; f32_to_bf16_k<<<(nc + 255) / 256, 256, 0, stream>>>(in_proj_w, wqkv, nc);
    nc = EMB * EMB;     f32_to_bf16_k<<<(nc + 255) / 256, 256, 0, stream>>>(out_w, wout, nc);
    nc = 4 * EMB * EMB; f32_to_bf16_k<<<(nc + 255) / 256, 256, 0, stream>>>(fc_w, wfc, nc);
    nc = 4 * EMB * EMB; f32_to_bf16_k<<<(nc + 255) / 256, 256, 0, stream>>>(proj_w, wproj, nc);

    // y = LN1(x) -> bf16
    ln_to_bf16_k<<<MROWS, 256, 0, stream>>>(x, ln1_w, ln1_b, y_bf);
    // qkv = y * Wqkv^T + b  (4096 x 3072 x 1024)
    gemm_bf16_k<false, false, false>
        <<<dim3(3 * EMB / 64, MROWS / 256), 256, 0, stream>>>(
            y_bf, wqkv, in_proj_b, nullptr, qkv_f, nullptr, MROWS, 3 * EMB, EMB);
    // split/scale/transpose into head layouts
    pack_qkv_k<<<MROWS, 256, 0, stream>>>(qkv_f, q_p, k_p, v_t);
    // attention (flash, WMMA)
    flash_attn_k<<<dim3(NB * NHEAD, L_SEQ / 128), 256, 0, stream>>>(q_p, k_p, v_t, o_bf);
    // x1 = x + o * Wout^T + b
    gemm_bf16_k<false, true, false>
        <<<dim3(EMB / 64, MROWS / 256), 256, 0, stream>>>(
            o_bf, wout, out_b, x, x1, nullptr, MROWS, EMB, EMB);
    // z = LN2(x1) -> bf16
    ln_to_bf16_k<<<MROWS, 256, 0, stream>>>(x1, ln2_w, ln2_b, z_bf);
    // h = GELU(z * Wfc^T + b) -> bf16  (4096 x 4096 x 1024)
    gemm_bf16_k<true, false, true>
        <<<dim3(4 * EMB / 64, MROWS / 256), 256, 0, stream>>>(
            z_bf, wfc, fc_b, nullptr, nullptr, h_bf, MROWS, 4 * EMB, EMB);
    // out = x1 + h * Wproj^T + b  (4096 x 1024 x 4096)
    gemm_bf16_k<false, true, false>
        <<<dim3(EMB / 64, MROWS / 256), 256, 0, stream>>>(
            h_bf, wproj, proj_b, x1, (float*)d_out, nullptr, MROWS, EMB, 4 * EMB);
}
